// HeteroGNN_8521215115545
// MI455X (gfx1250) — matrix operand order repaired and checked by reading, not compiled, not run on hardware
//
#include <hip/hip_runtime.h>

// ---------------------------------------------------------------------------
// HeteroGNN (2-layer hetero-GAT + MLP decoder) for MI455X / gfx1250.
//
// Dense GEMMs: v_wmma_f32_16x16x32_bf16, fp32 accumulate.
//   - B (weights) pre-packed once per GEMM into bf16 WMMA-fragment layout
//     (each lane's 16 bf16 values contiguous -> two global_load_b128).
//   - A (fp32 activations) loaded as float4 (global_load_b128), packed to
//     bf16 in registers (2 VALU per dword).
//   - Decoder h1 (the traffic heavyweight: 150000x512) is stored in bf16 and
//     consumed by a bf16-A GEMM variant: two b128 loads per fragment, zero
//     packing VALU. Saves ~300 MB (~30%) of total DRAM traffic.
//   - 4 output N-tiles per wave (2 when N==32): one A fragment feeds 4 WMMAs.
//   - Decoder layer 1 fuses concat(xd[src], xp[dst]) into the A load; the
//     source select is resolved at compile time per unrolled K-tile.
// Segment softmax over edges: float atomics (resolved in the 192MB L2).
//
// Workspace: ~330 MB (h1 now bf16).
// ---------------------------------------------------------------------------

typedef __attribute__((ext_vector_type(16))) __bf16    bf16x16;
typedef __attribute__((ext_vector_type(8)))  float     f32x8;
typedef __attribute__((ext_vector_type(4)))  float     f32x4;
typedef __attribute__((ext_vector_type(8)))  unsigned  u32x8;
typedef __attribute__((ext_vector_type(4)))  unsigned  u32x4;

constexpr int kH    = 4;
constexpr int kC    = 32;
constexpr int kHC   = 128;     // H*C
constexpr int kND   = 30000;   // #drug nodes    (multiple of 16)
constexpr int kNP   = 20000;   // #protein nodes (multiple of 16)
constexpr int kEint = 150000;  // int / rev edges
constexpr int kEdd  = 120000;
constexpr int kEpp  = 100000;

// pack two fp32 into a dword of two bf16 (truncating round)
__device__ __forceinline__ unsigned pk2bf(float lo, float hi) {
  return (__float_as_uint(lo) >> 16) | (__float_as_uint(hi) & 0xFFFF0000u);
}

// Build a lane's A fragment (ISA 7.12.2: elements 0..7 = K ak..ak+7,
// elements 8..15 = K ak+16..ak+23) from a row-major fp32 row pointer.
__device__ __forceinline__ bf16x16 loadA_frag(const float* __restrict__ rowp, int ak) {
  f32x4 a0 = *(const f32x4*)(rowp + ak);
  f32x4 a1 = *(const f32x4*)(rowp + ak + 4);
  f32x4 a2 = *(const f32x4*)(rowp + ak + 16);
  f32x4 a3 = *(const f32x4*)(rowp + ak + 20);
  u32x8 r;
  r[0] = pk2bf(a0[0], a0[1]); r[1] = pk2bf(a0[2], a0[3]);
  r[2] = pk2bf(a1[0], a1[1]); r[3] = pk2bf(a1[2], a1[3]);
  r[4] = pk2bf(a2[0], a2[1]); r[5] = pk2bf(a2[2], a2[3]);
  r[6] = pk2bf(a3[0], a3[1]); r[7] = pk2bf(a3[2], a3[3]);
  return __builtin_bit_cast(bf16x16, r);
}

// Same fragment from a row-major bf16 row: two b128 loads, no VALU packing.
__device__ __forceinline__ bf16x16 loadA_frag_bf16(const unsigned short* __restrict__ rowp,
                                                   int ak) {
  u32x4 lo = *(const u32x4*)(rowp + ak);        // K = ak   .. ak+7
  u32x4 hi = *(const u32x4*)(rowp + ak + 16);   // K = ak+16.. ak+23
  u32x8 r;
  r[0] = lo[0]; r[1] = lo[1]; r[2] = lo[2]; r[3] = lo[3];
  r[4] = hi[0]; r[5] = hi[1]; r[6] = hi[2]; r[7] = hi[3];
  return __builtin_bit_cast(bf16x16, r);
}

// sign-split float atomic max (memory initialized to -inf)
__device__ __forceinline__ void atomicMaxF(float* addr, float val) {
  if (val >= 0.0f) atomicMax((int*)addr, __float_as_int(val));
  else             atomicMin((unsigned int*)addr, __float_as_uint(val));
}

// ---------------------------------------------------------------------------
// Pack B[K,N] fp32 -> bf16 fragment layout: dword index
//   ((kt*tilesN + nt)*32 + lane)*8 + d,  lane={hi,hl}, k=kt*32+hi*8+(d>>2)*16+2*(d&3), n=nt*16+hl
// ---------------------------------------------------------------------------
__global__ void pack_b_bf16(const float* __restrict__ B, unsigned* __restrict__ Bp,
                            int N, int K) {
  int t = blockIdx.x * blockDim.x + threadIdx.x;
  const int total = (K * N) >> 1;
  if (t >= total) return;
  const int d      = t & 7;
  const int lane   = (t >> 3) & 31;
  const int tilesN = N >> 4;
  const int nt     = (t >> 8) % tilesN;
  const int kt     = (t >> 8) / tilesN;
  const int hl = lane & 15, hi = lane >> 4;
  const int k = (kt << 5) + (hi << 3) + ((d & 4) << 2) + ((d & 3) << 1);
  const int n = (nt << 4) + hl;
  Bp[t] = pk2bf(B[(size_t)k * N + n], B[(size_t)(k + 1) * N + n]);
}

// ---------------------------------------------------------------------------
// Plain WMMA GEMM: Y[M,N] = act(A[M,K] @ B + bias), B pre-packed bf16.
// One wave owns TPW 16x16 output tiles sharing one A fragment per K-step.
// ACT: 0 = none, 1 = relu.
// ---------------------------------------------------------------------------
template <int ACT, int TPW>
__global__ void gemm_wmma_kernel(const float* __restrict__ A,
                                 const unsigned* __restrict__ Bp,
                                 const float* __restrict__ bias,
                                 float* __restrict__ Y,
                                 int M, int N, int K) {
  const int lane    = threadIdx.x & 31;
  const int wave    = threadIdx.x >> 5;
  const int tilesN  = N >> 4;
  const int groupsN = tilesN / TPW;
  const int gid     = blockIdx.x * (blockDim.x >> 5) + wave;
  if (gid >= (M >> 4) * groupsN) return;

  const int tm  = (gid / groupsN) << 4;
  const int nt0 = (gid % groupsN) * TPW;
  const int hl = lane & 15, hi = lane >> 4;
  const float* rowp = A + (size_t)(tm + hl) * K;

  f32x8 acc[TPW];
#pragma unroll
  for (int t = 0; t < TPW; ++t) acc[t] = {};

  const int KT = K >> 5;
  for (int kt = 0; kt < KT; ++kt) {
    if (kt + 1 < KT) __builtin_prefetch(rowp + ((kt + 1) << 5), 0, 0);
    const bf16x16 a = loadA_frag(rowp, (kt << 5) + hi * 8);
    const unsigned* bbase = Bp + (((size_t)kt * tilesN + nt0) * 32 + lane) * 8;
#pragma unroll
    for (int t = 0; t < TPW; ++t) {
      const bf16x16 b = __builtin_bit_cast(bf16x16, *(const u32x8*)(bbase + (size_t)t * 256));
      acc[t] = __builtin_amdgcn_wmma_f32_16x16x32_bf16(
          false, a, false, b, (short)0, acc[t], false, false);
    }
  }

#pragma unroll
  for (int t = 0; t < TPW; ++t) {
    const int tn = (nt0 + t) << 4;
    const float bb = bias ? bias[tn + hl] : 0.0f;
#pragma unroll
    for (int r = 0; r < 8; ++r) {
      float v = acc[t][r] + bb;
      if (ACT == 1) v = fmaxf(v, 0.0f);
      Y[(size_t)(tm + hi * 8 + r) * N + tn + hl] = v;
    }
  }
}

// ---------------------------------------------------------------------------
// bf16-A WMMA GEMM (decoder layer 2): A is row-major bf16; fragment loads are
// two b128s with zero packing VALU. Output fp32, relu.
// ---------------------------------------------------------------------------
template <int TPW>
__global__ void gemm_a16_wmma_kernel(const unsigned short* __restrict__ A,
                                     const unsigned* __restrict__ Bp,
                                     const float* __restrict__ bias,
                                     float* __restrict__ Y,
                                     int M, int N, int K) {
  const int lane    = threadIdx.x & 31;
  const int wave    = threadIdx.x >> 5;
  const int tilesN  = N >> 4;
  const int groupsN = tilesN / TPW;
  const int gid     = blockIdx.x * (blockDim.x >> 5) + wave;
  if (gid >= (M >> 4) * groupsN) return;

  const int tm  = (gid / groupsN) << 4;
  const int nt0 = (gid % groupsN) * TPW;
  const int hl = lane & 15, hi = lane >> 4;
  const unsigned short* rowp = A + (size_t)(tm + hl) * K;

  f32x8 acc[TPW];
#pragma unroll
  for (int t = 0; t < TPW; ++t) acc[t] = {};

  const int KT = K >> 5;
  for (int kt = 0; kt < KT; ++kt) {
    if (kt + 1 < KT) __builtin_prefetch(rowp + ((kt + 1) << 5), 0, 0);
    const bf16x16 a = loadA_frag_bf16(rowp, (kt << 5) + hi * 8);
    const unsigned* bbase = Bp + (((size_t)kt * tilesN + nt0) * 32 + lane) * 8;
#pragma unroll
    for (int t = 0; t < TPW; ++t) {
      const bf16x16 b = __builtin_bit_cast(bf16x16, *(const u32x8*)(bbase + (size_t)t * 256));
      acc[t] = __builtin_amdgcn_wmma_f32_16x16x32_bf16(
          false, a, false, b, (short)0, acc[t], false, false);
    }
  }

#pragma unroll
  for (int t = 0; t < TPW; ++t) {
    const int tn = (nt0 + t) << 4;
    const float bb = bias[tn + hl];
#pragma unroll
    for (int r = 0; r < 8; ++r) {
      float v = fmaxf(acc[t][r] + bb, 0.0f);  // relu
      Y[(size_t)(tm + hi * 8 + r) * N + tn + hl] = v;
    }
  }
}

// ---------------------------------------------------------------------------
// Concat-gather WMMA GEMM for decoder layer 1: row r of A is
// concat(Xa[ga[r],0:128], Xb[gb[r],0:128]); K = 256, relu, TPW = 4.
// The Xa/Xb select is per-K-tile and resolved at compile time by full unroll.
// Output is written as row-major bf16 (halves h1 traffic for layer 2).
// ---------------------------------------------------------------------------
__global__ void gemm_cat_wmma_kernel(const float* __restrict__ Xa,
                                     const float* __restrict__ Xb,
                                     const int* __restrict__ ga,
                                     const int* __restrict__ gb,
                                     const unsigned* __restrict__ Bp,
                                     const float* __restrict__ bias,
                                     unsigned short* __restrict__ Y,  // bf16 out
                                     int M, int N) {
  constexpr int TPW = 4;
  const int lane    = threadIdx.x & 31;
  const int wave    = threadIdx.x >> 5;
  const int tilesN  = N >> 4;
  const int groupsN = tilesN / TPW;
  const int gid     = blockIdx.x * (blockDim.x >> 5) + wave;
  if (gid >= (M >> 4) * groupsN) return;

  const int tm  = (gid / groupsN) << 4;
  const int nt0 = (gid % groupsN) * TPW;
  const int hl = lane & 15, hi = lane >> 4;
  const float* rowp1 = Xa + (size_t)ga[tm + hl] * kHC;  // K in [0,128)
  const float* rowp2 = Xb + (size_t)gb[tm + hl] * kHC;  // K in [128,256)

  f32x8 acc[TPW];
#pragma unroll
  for (int t = 0; t < TPW; ++t) acc[t] = {};

#pragma unroll
  for (int kt = 0; kt < 8; ++kt) {  // K = 256 -> 8 K-tiles of 32
    const float* rp = (kt < 4) ? rowp1 : rowp2;         // compile-time select
    const bf16x16 a = loadA_frag(rp, ((kt & 3) << 5) + hi * 8);
    const unsigned* bbase = Bp + (((size_t)kt * tilesN + nt0) * 32 + lane) * 8;
#pragma unroll
    for (int t = 0; t < TPW; ++t) {
      const bf16x16 b = __builtin_bit_cast(bf16x16, *(const u32x8*)(bbase + (size_t)t * 256));
      acc[t] = __builtin_amdgcn_wmma_f32_16x16x32_bf16(
          false, a, false, b, (short)0, acc[t], false, false);
    }
  }

#pragma unroll
  for (int t = 0; t < TPW; ++t) {
    const int tn = (nt0 + t) << 4;
    const float bb = bias[tn + hl];
#pragma unroll
    for (int r = 0; r < 8; ++r) {
      float v = fmaxf(acc[t][r] + bb, 0.0f);  // relu
      Y[(size_t)(tm + hi * 8 + r) * N + tn + hl] =
          (unsigned short)(__float_as_uint(v) >> 16);  // bf16 store
    }
  }
}

// ------------------------- elementwise / edge kernels ----------------------

__global__ void fill_f32(float* __restrict__ p, float v, int n) {
  int t = blockIdx.x * blockDim.x + threadIdx.x;
  if (t < n) p[t] = v;
}

__global__ void init_m_den(float* __restrict__ m, float* __restrict__ den, int n) {
  int t = blockIdx.x * blockDim.x + threadIdx.x;
  if (t < n) { m[t] = -__builtin_inff(); den[t] = 0.0f; }
}

// a_head[i,h] = sum_c h[i, h*32+c] * att[h*32+c]
__global__ void attn_scores(const float* __restrict__ hmat,
                            const float* __restrict__ att,
                            float* __restrict__ out, int n) {
  int t = blockIdx.x * blockDim.x + threadIdx.x;
  if (t >= n * kH) return;
  const int i  = t >> 2;
  const int hh = t & 3;
  const float* row = hmat + (size_t)i * kHC + hh * kC;
  const float* av  = att + hh * kC;
  float s = 0.0f;
#pragma unroll
  for (int c = 0; c < kC; ++c) s += row[c] * av[c];
  out[t] = s;
}

// e = leaky(a_s[src]+a_d[dst], 0.2); segment max by dst (atomic)
__global__ void edge_score_max(const int* __restrict__ src, const int* __restrict__ dst,
                               const float* __restrict__ a_s, const float* __restrict__ a_d,
                               float* __restrict__ e_scr, float* __restrict__ m, int E) {
  int t = blockIdx.x * blockDim.x + threadIdx.x;
  if (t >= E * kH) return;
  const int e  = t >> 2;
  const int hh = t & 3;
  float v = a_s[src[e] * kH + hh] + a_d[dst[e] * kH + hh];
  v = v > 0.0f ? v : 0.2f * v;
  e_scr[t] = v;
  atomicMaxF(&m[dst[e] * kH + hh], v);
}

// m = isfinite(m) ? m : 0   (m is -inf only for empty segments)
__global__ void fix_m(float* __restrict__ m, int n) {
  int t = blockIdx.x * blockDim.x + threadIdx.x;
  if (t < n) { float v = m[t]; if (!(v >= -3.0e38f)) m[t] = 0.0f; }
}

// ex = exp(e - m[dst]); segment sum by dst (atomic)
__global__ void edge_exp_sum(const int* __restrict__ dst,
                             const float* __restrict__ e_scr, const float* __restrict__ m,
                             float* __restrict__ ex_scr, float* __restrict__ den, int E) {
  int t = blockIdx.x * blockDim.x + threadIdx.x;
  if (t >= E * kH) return;
  const int e  = t >> 2;
  const int hh = t & 3;
  float ex = expf(e_scr[t] - m[dst[e] * kH + hh]);
  ex_scr[t] = ex;
  atomicAdd(&den[dst[e] * kH + hh], ex);
}

// out[dst, :] += hs[src, :] * alpha   (one thread per (edge, feature))
__global__ void edge_scatter(const int* __restrict__ src, const int* __restrict__ dst,
                             const float* __restrict__ hs,
                             const float* __restrict__ ex_scr,
                             const float* __restrict__ den,
                             float* __restrict__ out, int E) {
  int t = blockIdx.x * blockDim.x + threadIdx.x;
  if (t >= E * kHC) return;
  const int e  = t >> 7;
  const int j  = t & 127;
  const int hh = j >> 5;
  const int s = src[e], d = dst[e];
  const float alpha = ex_scr[e * kH + hh] / (den[d * kH + hh] + 1e-16f);
  atomicAdd(&out[(size_t)d * kHC + j], hs[(size_t)s * kHC + j] * alpha);
}

// out = leaky(out + b0 + b1 + b2, 0.01)  — fold the 3 relation biases
__global__ void finalize_nodes(float* __restrict__ out,
                               const float* __restrict__ b0,
                               const float* __restrict__ b1,
                               const float* __restrict__ b2, int n) {
  int t = blockIdx.x * blockDim.x + threadIdx.x;
  if (t >= n * kHC) return;
  const int c = t & 127;
  float v = out[t] + b0[c] + b1[c] + b2[c];
  out[t] = v > 0.0f ? v : 0.01f * v;
}

// final 128 -> 1 projection
__global__ void final_dot(const float* __restrict__ h2, const float* __restrict__ W3,
                          const float* __restrict__ b3, float* __restrict__ out, int M) {
  int i = blockIdx.x * blockDim.x + threadIdx.x;
  if (i >= M) return;
  const float* row = h2 + (size_t)i * kHC;
  float s = 0.0f;
#pragma unroll
  for (int k = 0; k < kHC; ++k) s += row[k] * W3[k];
  out[i] = s + b3[0];
}

// ---------------------------------------------------------------------------
// Host orchestration
// ---------------------------------------------------------------------------

struct GatP { const float *Wd, *Ws, *ad, *as_, *b; };

static inline int blk(int n) { return (n + 255) / 256; }

static void gemm_launch(const float* A, const float* Bsrc, unsigned* bpack,
                        const float* bias, float* Y, int M, int N, int K, int act,
                        hipStream_t stream) {
  pack_b_bf16<<<blk((K * N) >> 1), 256, 0, stream>>>(Bsrc, bpack, N, K);
  const int tilesN = N >> 4;
  const int tpw    = (tilesN % 4 == 0) ? 4 : 2;
  const int groups = (M >> 4) * (tilesN / tpw);
  dim3 grid((groups + 3) / 4), tb(128);  // 4 wave32s per block
  if (tpw == 4) {
    if (act == 1) gemm_wmma_kernel<1, 4><<<grid, tb, 0, stream>>>(A, bpack, bias, Y, M, N, K);
    else          gemm_wmma_kernel<0, 4><<<grid, tb, 0, stream>>>(A, bpack, bias, Y, M, N, K);
  } else {
    if (act == 1) gemm_wmma_kernel<1, 2><<<grid, tb, 0, stream>>>(A, bpack, bias, Y, M, N, K);
    else          gemm_wmma_kernel<0, 2><<<grid, tb, 0, stream>>>(A, bpack, bias, Y, M, N, K);
  }
}

extern "C" void kernel_launch(void* const* d_in, const int* in_sizes, int n_in,
                              void* d_out, int out_size, void* d_ws, size_t ws_size,
                              hipStream_t stream) {
  (void)in_sizes; (void)n_in; (void)out_size; (void)ws_size;

  // --- inputs (setup_inputs dict order) ---
  const float* x_drug    = (const float*)d_in[0];   // 30000 x 1024
  const float* x_prot    = (const float*)d_in[1];   // 20000 x 1280
  const int*   ei_int    = (const int*)d_in[2];     // 2 x 150000
  const int*   ei_rev    = (const int*)d_in[3];     // 2 x 150000
  const int*   ei_sim_dd = (const int*)d_in[4];     // 2 x 120000
  const int*   ei_sem_dd = (const int*)d_in[5];     // 2 x 120000
  const int*   ei_sim_pp = (const int*)d_in[6];     // 2 x 100000
  const int*   ei_sem_pp = (const int*)d_in[7];     // 2 x 100000

  // --- params: JAX pytree flatten order (dict keys sorted) ---
  // top: conv1, conv2, dec, drug_lin, protein_lin
  // relations sorted: int, rev, sem_dd, sem_pp, sim_dd, sim_pp
  // per-GAT sorted: Wd, Ws, ad, as, b
  const int CONV1 = 8, CONV2 = 38;
  enum { R_INT = 0, R_REV = 1, R_SEM_DD = 2, R_SEM_PP = 3, R_SIM_DD = 4, R_SIM_PP = 5 };
  auto gat = [&](int base, int rel) -> GatP {
    const int i = base + rel * 5;
    return GatP{ (const float*)d_in[i + 0], (const float*)d_in[i + 1],
                 (const float*)d_in[i + 2], (const float*)d_in[i + 3],
                 (const float*)d_in[i + 4] };
  };
  const float* decW1  = (const float*)d_in[68];  // 256 x 512
  const float* decW2  = (const float*)d_in[69];  // 512 x 128
  const float* decW3  = (const float*)d_in[70];  // 128 x 1
  const float* decb1  = (const float*)d_in[71];
  const float* decb2  = (const float*)d_in[72];
  const float* decb3  = (const float*)d_in[73];
  const float* WdrugL = (const float*)d_in[74];  // 1024 x 32
  const float* bdrugL = (const float*)d_in[75];
  const float* WprotL = (const float*)d_in[76];  // 1280 x 32
  const float* bprotL = (const float*)d_in[77];

  // --- workspace carve (all counts are multiples of 8 -> 32B alignment kept)
  float* w = (float*)d_ws;
  auto carve = [&](size_t n) { float* p = w; w += n; return p; };
  float* xd0 = carve((size_t)kND * kC);
  float* xp0 = carve((size_t)kNP * kC);
  float* xd1 = carve((size_t)kND * kHC);
  float* xp1 = carve((size_t)kNP * kHC);
  float* xd2 = carve((size_t)kND * kHC);
  float* xp2 = carve((size_t)kNP * kHC);
  float* hs  = carve((size_t)kND * kHC);
  float* hd  = carve((size_t)kND * kHC);
  float* a_s = carve((size_t)kND * kH);
  float* a_d = carve((size_t)kND * kH);
  float* mM  = carve((size_t)kND * kH);
  float* den = carve((size_t)kND * kH);
  float* e_s = carve((size_t)kEint * kH);
  float* e_x = carve((size_t)kEint * kH);
  unsigned short* h1b = (unsigned short*)carve((size_t)kEint * 256);  // bf16 150000x512
  float* h2  = carve((size_t)kEint * kHC);
  unsigned* bpack = (unsigned*)carve(65536);  // max K*N/2 dwords (W1: 256*512/2)

  auto gemm = [&](const float* A, const float* B, const float* bias, float* Y,
                  int M, int N, int K, int act) {
    gemm_launch(A, B, bpack, bias, Y, M, N, K, act, stream);
  };

  // One GAT relation: projections (WMMA) + segment softmax + scatter.
  auto run_gat = [&](const float* xs, int ns, const float* xdv, int nd,
                     const int* ei, int E, GatP g, int din, float* accum) {
    gemm(xs,  g.Ws, nullptr, hs, ns, kHC, din, 0);
    gemm(xdv, g.Wd, nullptr, hd, nd, kHC, din, 0);
    attn_scores<<<blk(ns * kH), 256, 0, stream>>>(hs, g.as_, a_s, ns);
    attn_scores<<<blk(nd * kH), 256, 0, stream>>>(hd, g.ad,  a_d, nd);
    init_m_den<<<blk(nd * kH), 256, 0, stream>>>(mM, den, nd * kH);
    const int* src = ei;
    const int* dst = ei + E;
    edge_score_max<<<blk(E * kH), 256, 0, stream>>>(src, dst, a_s, a_d, e_s, mM, E);
    fix_m<<<blk(nd * kH), 256, 0, stream>>>(mM, nd * kH);
    edge_exp_sum<<<blk(E * kH), 256, 0, stream>>>(dst, e_s, mM, e_x, den, E);
    edge_scatter<<<blk(E * kHC), 256, 0, stream>>>(src, dst, hs, e_x, den, accum, E);
  };

  // ---- input linear layers (WMMA) ----
  gemm(x_drug, WdrugL, bdrugL, xd0, kND, kC, 1024, 0);
  gemm(x_prot, WprotL, bprotL, xp0, kNP, kC, 1280, 0);

  // ---- two hetero-GAT layers ----
  const float* xd_in = xd0; const float* xp_in = xp0;
  int din = kC;
  float* xd_o = xd1; float* xp_o = xp1;
  for (int layer = 0; layer < 2; ++layer) {
    const int base = (layer == 0) ? CONV1 : CONV2;
    fill_f32<<<blk(kND * kHC), 256, 0, stream>>>(xd_o, 0.0f, kND * kHC);
    fill_f32<<<blk(kNP * kHC), 256, 0, stream>>>(xp_o, 0.0f, kNP * kHC);

    // drug side: rev (prot->drug) + sim_dd + sem_dd
    GatP grev = gat(base, R_REV), gsdd = gat(base, R_SIM_DD), gedd = gat(base, R_SEM_DD);
    run_gat(xp_in, kNP, xd_in, kND, ei_rev,    kEint, grev, din, xd_o);
    run_gat(xd_in, kND, xd_in, kND, ei_sim_dd, kEdd,  gsdd, din, xd_o);
    run_gat(xd_in, kND, xd_in, kND, ei_sem_dd, kEdd,  gedd, din, xd_o);
    finalize_nodes<<<blk(kND * kHC), 256, 0, stream>>>(xd_o, grev.b, gsdd.b, gedd.b, kND);

    // protein side: int (drug->prot) + sim_pp + sem_pp
    GatP gint = gat(base, R_INT), gspp = gat(base, R_SIM_PP), gepp = gat(base, R_SEM_PP);
    run_gat(xd_in, kND, xp_in, kNP, ei_int,    kEint, gint, din, xp_o);
    run_gat(xp_in, kNP, xp_in, kNP, ei_sim_pp, kEpp,  gspp, din, xp_o);
    run_gat(xp_in, kNP, xp_in, kNP, ei_sem_pp, kEpp,  gepp, din, xp_o);
    finalize_nodes<<<blk(kNP * kHC), 256, 0, stream>>>(xp_o, gint.b, gspp.b, gepp.b, kNP);

    xd_in = xd_o; xp_in = xp_o; din = kHC;
    xd_o = xd2; xp_o = xp2;
  }

  // ---- decoder ----
  // layer 1: A = concat(xd2[ei_int[0]], xp2[ei_int[1]]) fused into A load;
  // output h1 written as bf16.
  pack_b_bf16<<<blk((256 * 512) >> 1), 256, 0, stream>>>(decW1, bpack, 512, 256);
  {
    const int groups = (kEint >> 4) * (512 >> 4) / 4;
    gemm_cat_wmma_kernel<<<dim3((groups + 3) / 4), 128, 0, stream>>>(
        xd_in, xp_in, ei_int, ei_int + kEint, bpack, decb1, h1b, kEint, 512);
  }
  // layer 2: 512 -> 128 on bf16-A path, relu
  pack_b_bf16<<<blk((512 * kHC) >> 1), 256, 0, stream>>>(decW2, bpack, kHC, 512);
  {
    const int groups = (kEint >> 4) * ((kHC >> 4) / 4);
    gemm_a16_wmma_kernel<4><<<dim3((groups + 3) / 4), 128, 0, stream>>>(
        h1b, bpack, decb2, h2, kEint, kHC, 512);
  }
  // layer 3: 128 -> 1
  final_dot<<<blk(kEint), 256, 0, stream>>>(h2, decW3, decb3, (float*)d_out, kEint);
}
